// BidirectionalMambaBlock_41171556499479
// MI455X (gfx1250) — matrix operand order repaired and checked by reading, not compiled, not run on hardware
//
#include <hip/hip_runtime.h>
#include <hip/hip_bf16.h>
#include <cstdint>

// ---------------------------------------------------------------------------
// BidirectionalMambaBlock for MI455X (gfx1250), wave32 + WMMA bf16.
//   B=8, L=4096, DIM=512, D_INNER=1024, D_STATE=16, DT_RANK=32, D_CONV=4
//   M = B*L = 32768 tokens.
// GEMMs: double-buffered LDS filled by CDNA5 async global->LDS DMA
// (global_load_async_to_lds_b128 / s_wait_asynccnt, no VGPR staging),
// 128x128 (or 128x64) block tiles, each wave owns WM x WN 16x16 WMMA tiles,
// fragments read as ds_load_b128 pairs.
// ---------------------------------------------------------------------------

typedef __attribute__((ext_vector_type(16))) __bf16 v16bf;
typedef __attribute__((ext_vector_type(8)))  float  v8f;

#define DIMc    512
#define DINNER  1024
#define DSTATE  16
#define LLEN    4096
#define BSZ     8
#define MTOK    (BSZ * LLEN)

// CDNA5 async global->LDS copy: per-lane 16B, LDS dest addr in VGPR,
// 64-bit global addr in VGPR pair. Tracked by ASYNCcnt.
__device__ __forceinline__ void async_b128_to_lds(unsigned lds_addr,
                                                  const void* gaddr) {
  asm volatile("global_load_async_to_lds_b128 %0, %1, off"
               :
               : "v"(lds_addr), "v"(gaddr)
               : "memory");
}
__device__ __forceinline__ void async_wait0() {
  asm volatile("s_wait_asynccnt 0x0" ::: "memory");
}

// ---------------------------------------------------------------------------
// f32 -> bf16 conversion (weights)
// ---------------------------------------------------------------------------
__global__ __launch_bounds__(256) void f32_to_bf16_kernel(
    const float* __restrict__ in, __bf16* __restrict__ out, long n) {
  long i = (long)blockIdx.x * 256 + threadIdx.x;
  if (i < n) out[i] = (__bf16)in[i];
}

// ---------------------------------------------------------------------------
// LayerNorm over last dim (512), writes bf16. One block per token.
// ---------------------------------------------------------------------------
__global__ __launch_bounds__(256) void layernorm_kernel(
    const float* __restrict__ x, const float* __restrict__ gamma,
    const float* __restrict__ beta, __bf16* __restrict__ out) {
  const long token = blockIdx.x;
  const float* xp = x + token * DIMc;
  const int tid = threadIdx.x;
  float a0 = xp[tid], a1 = xp[tid + 256];
  float s = a0 + a1, ss = a0 * a0 + a1 * a1;
#pragma unroll
  for (int off = 16; off > 0; off >>= 1) {
    s  += __shfl_down(s,  off, 32);
    ss += __shfl_down(ss, off, 32);
  }
  __shared__ float rs[8], rss[8], stat[2];
  const int wave = tid >> 5, lane = tid & 31;
  if (lane == 0) { rs[wave] = s; rss[wave] = ss; }
  __syncthreads();
  if (tid == 0) {
    float ts = 0.f, tss = 0.f;
    for (int i = 0; i < 8; ++i) { ts += rs[i]; tss += rss[i]; }
    float mu  = ts * (1.f / DIMc);
    float var = tss * (1.f / DIMc) - mu * mu;
    stat[0] = mu;
    stat[1] = rsqrtf(var + 1e-5f);
  }
  __syncthreads();
  float mu = stat[0], rstd = stat[1];
  out[token * DIMc + tid]       = (__bf16)((a0 - mu) * rstd * gamma[tid]       + beta[tid]);
  out[token * DIMc + tid + 256] = (__bf16)((a1 - mu) * rstd * gamma[tid + 256] + beta[tid + 256]);
}

// ---------------------------------------------------------------------------
// WMMA GEMM: C[M,N] = act(A[M,K](bf16) * W[N,K]^T(bf16) [+bias]) [+resid]
// 256 threads = 8 waves (WROWS x WCOLS); each wave owns WM x WN 16x16 tiles.
// K stepped by 32; double-buffered LDS filled by async global->LDS b128 DMA
// overlapping the WMMAs; s_wait_asynccnt + one barrier per K-step.
// LDS row stride 40 elems -> every fragment piece is one ds_load_b128.
// MODE bits: 1=bias, 2=resid(f32), 4=outF(f32), 8=outB(bf16), 16=softplus.
// ---------------------------------------------------------------------------
template <int WROWS, int WCOLS, int WM, int WN, int MODE>
__global__ __launch_bounds__(256, 2) void wmma_gemm_kernel(
    const __bf16* __restrict__ A, int lda,
    const __bf16* __restrict__ W, int K,
    const float*  __restrict__ bias,
    const float*  __restrict__ resid,
    float*        __restrict__ outF,
    __bf16*       __restrict__ outB,
    int ldc) {
  constexpr bool HAS_BIAS = (MODE & 1)  != 0;
  constexpr bool HAS_RES  = (MODE & 2)  != 0;
  constexpr bool HAS_F    = (MODE & 4)  != 0;
  constexpr bool HAS_B    = (MODE & 8)  != 0;
  constexpr bool SOFTPLUS = (MODE & 16) != 0;

  constexpr int BM   = WROWS * WM * 16;
  constexpr int BN   = WCOLS * WN * 16;
  constexpr int STR  = 40;                    // LDS row stride, rows 16B-aligned
  constexpr int A_PT = (BM * 32 / 8) / 256;   // b128 chunks per thread (A tile)
  constexpr int W_PT = (BN * 32 / 8) / 256;   // b128 chunks per thread (W tile)

  __shared__ __bf16 sA[2][BM * STR];
  __shared__ __bf16 sW[2][BN * STR];

  const int tid  = threadIdx.x;
  const int lane = tid & 31;
  const int wave = tid >> 5;
  const int wRow = wave / WCOLS;
  const int wCol = wave % WCOLS;
  const long rowBase = (long)blockIdx.y * BM;
  const long colBase = (long)blockIdx.x * BN;

  v8f acc[WM][WN];
#pragma unroll
  for (int i = 0; i < WM; ++i)
#pragma unroll
    for (int j = 0; j < WN; ++j)
      acc[i][j] = (v8f){0.f, 0.f, 0.f, 0.f, 0.f, 0.f, 0.f, 0.f};

  // Issue async DMA of one K-tile straight into LDS buffer `buf`.
  auto issueTiles = [&](int k0, int buf) {
#pragma unroll
    for (int i = 0; i < A_PT; ++i) {
      int v = tid + i * 256;
      int row = v >> 2, cg = (v & 3) * 8;
      const __bf16* g = A + (rowBase + row) * (long)lda + k0 + cg;
      async_b128_to_lds((unsigned)(uintptr_t)&sA[buf][row * STR + cg], g);
      if (i == 0 && k0 + 32 < K)             // warm caches for tile after next
        __builtin_prefetch(g + 32, 0, 0);
    }
#pragma unroll
    for (int i = 0; i < W_PT; ++i) {
      int v = tid + i * 256;
      int row = v >> 2, cg = (v & 3) * 8;
      const __bf16* g = W + (colBase + row) * (long)K + k0 + cg;
      async_b128_to_lds((unsigned)(uintptr_t)&sW[buf][row * STR + cg], g);
    }
  };

  const int m  = lane & 15;
  const int hi = lane >> 4;
  const int nsteps = K / 32;

  issueTiles(0, 0);
  async_wait0();
  __syncthreads();

  for (int s = 0; s < nsteps; ++s) {
    const int buf = s & 1;
    if (s + 1 < nsteps) issueTiles((s + 1) * 32, buf ^ 1);

    // Fragments per CDNA5 ISA layouts; each half-fragment is one ds_load_b128.
    union U { uint4 q[2]; v16bf v; };
    U afr[WM], bfr[WN];
#pragma unroll
    for (int i = 0; i < WM; ++i) {
      const __bf16* base = &sA[buf][((wRow * WM + i) * 16 + m) * STR];
      afr[i].q[0] = *reinterpret_cast<const uint4*>(base + hi * 8);
      afr[i].q[1] = *reinterpret_cast<const uint4*>(base + 16 + hi * 8);
    }
#pragma unroll
    for (int j = 0; j < WN; ++j) {
      const __bf16* base = &sW[buf][((wCol * WN + j) * 16 + m) * STR];
      bfr[j].q[0] = *reinterpret_cast<const uint4*>(base + hi * 8);
      bfr[j].q[1] = *reinterpret_cast<const uint4*>(base + 16 + hi * 8);
    }
#pragma unroll
    for (int i = 0; i < WM; ++i)
#pragma unroll
      for (int j = 0; j < WN; ++j)
        acc[i][j] = __builtin_amdgcn_wmma_f32_16x16x32_bf16(
            false, afr[i].v, false, bfr[j].v, (short)0, acc[i][j], false, false);

    if (s + 1 < nsteps) {
      async_wait0();      // this wave's DMA into buf^1 landed
      __syncthreads();    // everyone done reading buf & their DMA done
    }
  }

  // Epilogue. C/D layout: col = lane&15, VGPR r -> row r + 8*(lane>>4).
  const int n     = lane & 15;
  const int rbase = (lane >> 4) * 8;
#pragma unroll
  for (int j = 0; j < WN; ++j) {
    const long col = colBase + (wCol * WN + j) * 16 + n;
    float bi = 0.f;
    if constexpr (HAS_BIAS) bi = bias[col];
#pragma unroll
    for (int i = 0; i < WM; ++i) {
      const long rowb = rowBase + (wRow * WM + i) * 16 + rbase;
#pragma unroll
      for (int r = 0; r < 8; ++r) {
        const long o = (rowb + r) * (long)ldc + col;
        float v = acc[i][j][r] + bi;
        if constexpr (SOFTPLUS) v = (v > 20.f) ? v : __logf(1.f + __expf(v));
        if constexpr (HAS_RES)  v += resid[o];
        if constexpr (HAS_F)    outF[o] = v;
        if constexpr (HAS_B)    outB[o] = (__bf16)v;
      }
    }
  }
}

// ---------------------------------------------------------------------------
// Depthwise causal conv (d_conv=4) + SiLU. reverse=0: look back (fwd branch),
// reverse=1: conv on time-reversed sequence == look forward, reversed taps.
// ---------------------------------------------------------------------------
__global__ __launch_bounds__(256) void dwconv_silu_kernel(
    const __bf16* __restrict__ xz, const float* __restrict__ w,
    const float* __restrict__ bconv, __bf16* __restrict__ xc, int reverse) {
  long idx = (long)blockIdx.x * 256 + threadIdx.x;   // over M*1024
  int  d  = (int)(idx & (DINNER - 1));
  long bt = idx >> 10;
  int  t  = (int)(bt % LLEN);
  long bb = bt / LLEN;
  float acc = bconv[d];
#pragma unroll
  for (int j = 0; j < 4; ++j) {
    int tt = reverse ? (t + 3 - j) : (t - 3 + j);
    if (tt >= 0 && tt < LLEN)
      acc += w[d * 4 + j] * (float)xz[(bb * LLEN + tt) * (2 * DINNER) + d];
  }
  acc = acc / (1.f + __expf(-acc));   // SiLU
  xc[idx] = (__bf16)acc;
}

// ---------------------------------------------------------------------------
// Selective scan: one thread per (batch, channel), 16-state recurrence in
// registers, B_t/C_t staged via LDS. Fused skip (+xc*D) and gate (*silu(z)).
// grid = (DINNER/256, BSZ)
// ---------------------------------------------------------------------------
__global__ __launch_bounds__(256) void scan_kernel(
    const float*  __restrict__ xdbl,   // (M,64): [dt_in(32) | B(16) | C(16)]
    const float*  __restrict__ dt,     // (M,1024) softplus'd
    const __bf16* __restrict__ xc,     // (M,1024)
    const __bf16* __restrict__ xz,     // (M,2048), z at col 1024+d
    const float*  __restrict__ A_log,  // (1024,16)
    const float*  __restrict__ Dp,     // (1024)
    __bf16*       __restrict__ yout,   // (M,1024)
    int reverse) {
  const int  d = blockIdx.x * 256 + threadIdx.x;
  const long b = blockIdx.y;
  float a[DSTATE], h[DSTATE];
#pragma unroll
  for (int s = 0; s < DSTATE; ++s) {
    a[s] = -__expf(A_log[d * DSTATE + s]);   // A = -exp(A_log)
    h[s] = 0.f;
  }
  const float Dv = Dp[d];
  __shared__ float sBC[2 * DSTATE];

  for (int i = 0; i < LLEN; ++i) {
    const int  t   = reverse ? (LLEN - 1 - i) : i;
    const long row = b * LLEN + t;
    __syncthreads();
    if (threadIdx.x < 2 * DSTATE)
      sBC[threadIdx.x] = xdbl[row * 64 + 32 + threadIdx.x];
    __syncthreads();
    const float dtv = dt[row * DINNER + d];
    const float xv  = (float)xc[row * DINNER + d];
    const float zv  = (float)xz[row * (2 * DINNER) + DINNER + d];
    const float dx  = dtv * xv;
    float y = 0.f;
#pragma unroll
    for (int s = 0; s < DSTATE; ++s) {
      float dA = __expf(dtv * a[s]);
      h[s] = h[s] * dA + dx * sBC[s];
      y += h[s] * sBC[DSTATE + s];
    }
    y += xv * Dv;
    y *= zv / (1.f + __expf(-zv));   // * silu(z)
    yout[row * DINNER + d] = (__bf16)y;
  }
}

// ---------------------------------------------------------------------------
// Host orchestration
// ---------------------------------------------------------------------------
static void cvt(const float* src, __bf16* dst, long n, hipStream_t s) {
  long blocks = (n + 255) / 256;
  f32_to_bf16_kernel<<<(unsigned)blocks, 256, 0, s>>>(src, dst, n);
}

extern "C" void kernel_launch(void* const* d_in, const int* in_sizes, int n_in,
                              void* d_out, int out_size, void* d_ws, size_t ws_size,
                              hipStream_t stream) {
  (void)in_sizes; (void)n_in; (void)out_size; (void)ws_size;
  const float* x      = (const float*)d_in[0];
  const float* norm_w = (const float*)d_in[1];
  const float* norm_b = (const float*)d_in[2];
  const float* cp_w   = (const float*)d_in[3];
  const float* cp_b   = (const float*)d_in[4];
  const float* fus_w  = (const float*)d_in[5];
  const float* fus_b  = (const float*)d_in[6];
  // per-direction: in_w, conv_w, conv_b, xproj_w, dt_w, dt_b, A_log, D, out_w
  const float* P[2][9];
  for (int dir = 0; dir < 2; ++dir)
    for (int j = 0; j < 9; ++j)
      P[dir][j] = (const float*)d_in[7 + dir * 9 + j];

  char* ws = (char*)d_ws;
  size_t off = 0;
  auto alloc = [&](size_t bytes) -> void* {
    off = (off + 255) & ~(size_t)255;
    void* p = ws + off;
    off += bytes;
    return p;
  };

  __bf16* cp_wb  = (__bf16*)alloc((size_t)DIMc * DIMc * 2);
  __bf16* fus_wb = (__bf16*)alloc((size_t)DIMc * DINNER * 2);
  __bf16 *in_wb[2], *xproj_wb[2], *dt_wb[2], *out_wb[2];
  for (int dir = 0; dir < 2; ++dir) {
    in_wb[dir]    = (__bf16*)alloc((size_t)(2 * DINNER) * DIMc * 2);
    xproj_wb[dir] = (__bf16*)alloc((size_t)64 * DINNER * 2);
    dt_wb[dir]    = (__bf16*)alloc((size_t)DINNER * 32 * 2);
    out_wb[dir]   = (__bf16*)alloc((size_t)DIMc * DINNER * 2);
  }
  __bf16* hn_b     = (__bf16*)alloc((size_t)MTOK * DIMc * 2);
  __bf16* h_b      = (__bf16*)alloc((size_t)MTOK * DIMc * 2);
  __bf16* xz_b     = (__bf16*)alloc((size_t)MTOK * 2 * DINNER * 2);
  __bf16* xc_b     = (__bf16*)alloc((size_t)MTOK * DINNER * 2);
  float*  xdbl_f   = (float*) alloc((size_t)MTOK * 64 * 4);
  __bf16* xdbl_b   = (__bf16*)alloc((size_t)MTOK * 64 * 2);
  float*  dt_f     = (float*) alloc((size_t)MTOK * DINNER * 4);
  __bf16* yg_b     = (__bf16*)alloc((size_t)MTOK * DINNER * 2);
  __bf16* concat_b = (__bf16*)alloc((size_t)MTOK * DINNER * 2);   // [fwd|bwd]

  // weight conversions
  cvt(cp_w,  cp_wb,  (long)DIMc * DIMc, stream);
  cvt(fus_w, fus_wb, (long)DIMc * DINNER, stream);
  for (int dir = 0; dir < 2; ++dir) {
    cvt(P[dir][0], in_wb[dir],    (long)(2 * DINNER) * DIMc, stream);
    cvt(P[dir][3], xproj_wb[dir], (long)64 * DINNER, stream);
    cvt(P[dir][4], dt_wb[dir],    (long)DINNER * 32, stream);
    cvt(P[dir][8], out_wb[dir],   (long)DIMc * DINNER, stream);
  }

  // LayerNorm -> hn (bf16)
  layernorm_kernel<<<MTOK, 256, 0, stream>>>(x, norm_w, norm_b, hn_b);

  // Main config: block 128x128, wave 32x64 (2x4 tiles).
  // Narrow config (N=64): block 128x64, wave 16x64 (1x4 tiles).
  // MODE bits: 1=bias, 2=resid, 4=outF, 8=outB, 16=softplus.

  // channel projection: h = hn @ cp_w^T + cp_b  -> bf16
  wmma_gemm_kernel<4, 2, 2, 4, 9><<<dim3(DIMc / 128, MTOK / 128), 256, 0, stream>>>(
      hn_b, DIMc, cp_wb, DIMc, cp_b, nullptr, nullptr, h_b, DIMc);

  for (int dir = 0; dir < 2; ++dir) {
    // xz = h @ in_w^T   (M x 2048)
    wmma_gemm_kernel<4, 2, 2, 4, 8><<<dim3(2 * DINNER / 128, MTOK / 128), 256, 0, stream>>>(
        h_b, DIMc, in_wb[dir], DIMc, nullptr, nullptr, nullptr, xz_b, 2 * DINNER);
    // depthwise causal conv + SiLU (direction-aware)
    dwconv_silu_kernel<<<(unsigned)((long)MTOK * DINNER / 256), 256, 0, stream>>>(
        xz_b, P[dir][1], P[dir][2], xc_b, dir);
    // x_dbl = xc @ xproj^T  (M x 64): f32 (B,C for scan) + bf16 (dt GEMM)
    wmma_gemm_kernel<8, 1, 1, 4, 12><<<dim3(1, MTOK / 128), 256, 0, stream>>>(
        xc_b, DINNER, xproj_wb[dir], DINNER, nullptr, nullptr, xdbl_f, xdbl_b, 64);
    // dt = softplus(x_dbl[:, :32] @ dt_w^T + dt_b)  (M x 1024)
    wmma_gemm_kernel<4, 2, 2, 4, 21><<<dim3(DINNER / 128, MTOK / 128), 256, 0, stream>>>(
        xdbl_b, 64, dt_wb[dir], 32, P[dir][5], nullptr, dt_f, nullptr, DINNER);
    // selective scan with fused skip + gate  -> yg (bf16)
    scan_kernel<<<dim3(DINNER / 256, BSZ), 256, 0, stream>>>(
        xdbl_f, dt_f, xc_b, xz_b, P[dir][6], P[dir][7], yg_b, dir);
    // out projection -> concat buffer column block [dir*512, dir*512+512)
    wmma_gemm_kernel<4, 2, 2, 4, 8><<<dim3(DIMc / 128, MTOK / 128), 256, 0, stream>>>(
        yg_b, DINNER, out_wb[dir], DINNER, nullptr, nullptr, nullptr,
        concat_b + dir * DIMc, DINNER);
  }

  // fused = concat @ fus_w^T + fus_b + x   -> d_out (f32)
  wmma_gemm_kernel<4, 2, 2, 4, 7><<<dim3(DIMc / 128, MTOK / 128), 256, 0, stream>>>(
      concat_b, DINNER, fus_wb, DINNER, fus_b, x, (float*)d_out, nullptr, DIMc);
}
